// FactorAtt_ConvRelPosEnc_85091892068736
// MI455X (gfx1250) — matrix operand order repaired and checked by reading, not compiled, not run on hardware
//
#include <hip/hip_runtime.h>
#include <hip/hip_bf16.h>

// ---------------------------------------------------------------------------
// FactorAtt (CoaT factorized attention) for MI455X / gfx1250, bf16 WMMA path.
// B=4, N=16384, C=512, H=8, D=64.
// ---------------------------------------------------------------------------

typedef __attribute__((ext_vector_type(16))) __bf16        v16bf;
typedef __attribute__((ext_vector_type(8)))  float         v8f;
typedef __attribute__((ext_vector_type(4)))  float         v4f;
typedef __attribute__((ext_vector_type(4)))  unsigned int  v4u;
typedef __attribute__((ext_vector_type(2)))  unsigned int  v2u;

union Frag16 {            // one WMMA 16-bit operand: 16 bf16 = 8 VGPRs = 32B/lane
    v16bf          v;
    unsigned short us[16];
    v4u            q[2];
};

__device__ __forceinline__ unsigned short f32_to_bf16(float f) {
    union { float f; unsigned int u; } x; x.f = f;
    unsigned int r = x.u + 0x7FFFu + ((x.u >> 16) & 1u);   // round-to-nearest-even
    return (unsigned short)(r >> 16);
}

__device__ __forceinline__ unsigned int pack2_bf16(float lo, float hi) {
    return (unsigned int)f32_to_bf16(lo) | ((unsigned int)f32_to_bf16(hi) << 16);
}

__device__ __forceinline__ v8f wmma_bf16(const Frag16& a, const Frag16& b, v8f c) {
    return __builtin_amdgcn_wmma_f32_16x16x32_bf16(
        false, a.v, false, b.v, (short)0, c, false, false);
}

// Problem constants
constexpr int B_ = 4, N_ = 16384, C_ = 512, H_ = 8, D_ = 64;
constexpr int M_ = B_ * N_;                 // 65536 rows for the big GEMMs
constexpr int NS = 8;                       // ktv N-reduction splits
constexpr int NBLK = M_ / 32;               // 2048 row-blocks of the GEMMs

// Workspace layout (bytes)
constexpr size_t WBF_SZ   = (size_t)C_ * C_ * 2;          // 512 KB bf16 weight
constexpr size_t OFF_WQ   = 0;
constexpr size_t OFF_WK   = OFF_WQ + WBF_SZ;
constexpr size_t OFF_WV   = OFF_WK + WBF_SZ;
constexpr size_t OFF_WP   = OFF_WV + WBF_SZ;
constexpr size_t QH_SZ    = (size_t)M_ * C_ * 2;          // 64 MB bf16
constexpr size_t OFF_QH   = OFF_WP + WBF_SZ;
constexpr size_t OFF_VH   = OFF_QH + QH_SZ;
constexpr size_t KH_SZ    = (size_t)M_ * C_ * 4;          // 128 MB f32
constexpr size_t OFF_KH   = OFF_VH + QH_SZ;
constexpr size_t MPART_SZ = (size_t)NBLK * C_ * 4;        // 4 MB col-max partials
constexpr size_t OFF_MP   = OFF_KH + KH_SZ;
constexpr size_t EP_SZ    = (size_t)B_ * H_ * NS * D_ * D_ * 4;  // 4 MB
constexpr size_t OFF_EP   = OFF_MP + MPART_SZ;
constexpr size_t ZP_SZ    = (size_t)B_ * H_ * NS * D_ * 4;
constexpr size_t OFF_ZP   = OFF_EP + EP_SZ;
constexpr size_t OFF_KTVT = OFF_ZP + ZP_SZ;               // [B,H,64(v),64(d)] bf16

// ---------------------------------------------------------------------------
// Kernel 0: f32 -> bf16 weight conversion.
// ---------------------------------------------------------------------------
__global__ void cvt_bf16_kernel(const float* __restrict__ src,
                                unsigned short* __restrict__ dst, int n) {
    int i = blockIdx.x * blockDim.x + threadIdx.x;
    if (i < n) dst[i] = f32_to_bf16(src[i]);
}

// ---------------------------------------------------------------------------
// Kernel 1: Y[M,512] = A[M,512] @ W.T  (A is f32, converted on the way in).
// Block = 32 rows x 512 cols (8 waves; wave tile 16M x 128N, 8 accumulators).
// A tile (32x32) staged once per K-step into LDS, software-pipelined; A is
// read from HBM exactly once. grid = (M/32).
// COLMAX: fuse per-block column-max of the (f32) output into the epilogue
// (feeds the softmax; kills a separate 128 MB scan of kh).
// ---------------------------------------------------------------------------
template <bool OUT_BF16, bool COLMAX>
__global__ __launch_bounds__(256) void
gemm_xwt_kernel(const float* __restrict__ Ap,
                const unsigned short* __restrict__ Wbf,
                void* __restrict__ Outp,
                float* __restrict__ Mpart) {
    constexpr int AS = 40;                      // LDS row stride (elements, 80B)
    __shared__ __align__(16) unsigned short Al[32 * AS];
    __shared__ float Mb[COLMAX ? 4 * 512 : 1];  // per-(mt,lq) column maxima

    const int tid = threadIdx.x;
    const int wv = tid >> 5, ln = tid & 31, lh = ln & 15, lq = ln >> 4;
    const int mt = wv & 1;                      // m-subtile (0/1)
    const int nt = wv >> 1;                     // n-subtile (0..3)
    const int mblk = blockIdx.x * 32;
    const int m0 = mblk + mt * 16;
    const int n0 = nt * 128;

    // staging role: thread t copies 4 elements of the 32x32 A tile
    const int srow = tid >> 3;                  // 0..31
    const int sck  = tid & 7;                   // 0..7 (4-element chunks)
    const size_t sgoff = (size_t)(mblk + srow) * C_ + sck * 4;
    unsigned short* sl = &Al[srow * AS + sck * 4];

    const v8f zero = {0.f, 0.f, 0.f, 0.f, 0.f, 0.f, 0.f, 0.f};
    v8f acc[8] = {zero, zero, zero, zero, zero, zero, zero, zero};

    v4f pf = *(const v4f*)(Ap + sgoff);         // in-flight staging data

    for (int ks = 0; ks < 16; ++ks) {
        __syncthreads();                        // everyone done reading Al
        {
            v2u st;
            st[0] = pack2_bf16(pf[0], pf[1]);
            st[1] = pack2_bf16(pf[2], pf[3]);
            *(v2u*)sl = st;
        }
        if (ks < 15) {                          // issue next tile's global load
            const size_t off = sgoff + (size_t)(ks + 1) * 32;
            pf = *(const v4f*)(Ap + off);
            __builtin_prefetch(Ap + off + 32, 0, 1);
        }
        __syncthreads();                        // staged tile visible

        const int k0 = ks * 32;
        Frag16 a;                               // A frag from LDS (shared by 8 tiles)
        {
            const unsigned short* ap = &Al[(mt * 16 + lh) * AS + lq * 16];
            a.q[0] = *(const v4u*)(ap);
            a.q[1] = *(const v4u*)(ap + 8);
        }
#pragma unroll
        for (int t = 0; t < 8; ++t) {           // 8 B frags -> 8 independent WMMA
            Frag16 b;
            const unsigned short* br =
                Wbf + (size_t)(n0 + t * 16 + lh) * C_ + k0 + lq * 16;
            b.q[0] = *(const v4u*)(br);
            b.q[1] = *(const v4u*)(br + 8);
            acc[t] = wmma_bf16(a, b, acc[t]);
        }
    }

    // store: row = m0 + r + 8*lq, col = n0 + 16t + lh
#pragma unroll
    for (int t = 0; t < 8; ++t) {
        const int col = n0 + t * 16 + lh;
#pragma unroll
        for (int r = 0; r < 8; ++r) {
            const size_t row = (size_t)(m0 + r + lq * 8);
            if (OUT_BF16)
                ((unsigned short*)Outp)[row * C_ + col] = f32_to_bf16(acc[t][r]);
            else
                ((float*)Outp)[row * C_ + col] = acc[t][r];
        }
    }

    if (COLMAX) {   // fused per-block column max (deterministic: pure fmax)
#pragma unroll
        for (int t = 0; t < 8; ++t) {
            const int col = n0 + t * 16 + lh;
            float mx = acc[t][0];
#pragma unroll
            for (int r = 1; r < 8; ++r) mx = fmaxf(mx, acc[t][r]);
            Mb[(mt * 2 + lq) * 512 + col] = mx;  // unique writer per slot
        }
        __syncthreads();
        for (int c = tid; c < 512; c += 256) {
            const float m = fmaxf(fmaxf(Mb[c], Mb[512 + c]),
                                  fmaxf(Mb[1024 + c], Mb[1536 + c]));
            Mpart[(size_t)blockIdx.x * 512 + c] = m;
        }
    }
}

// ---------------------------------------------------------------------------
// Kernel 2a: per (b,h,split): stream 2048 rows in 128-row chunks:
//   P = exp(kh - M) -> LDS transposed [d][n], vh -> LDS transposed [v][n],
//   WMMA-accumulate partial E[64,64] and partial Z[64]. 256 blocks total.
// ---------------------------------------------------------------------------
constexpr int PS = 136;                  // padded LDS row stride (elements)

__global__ __launch_bounds__(256) void
ktv_part_kernel(const float* __restrict__ kh,
                const unsigned short* __restrict__ vh,
                const float* __restrict__ Mpart,
                float* __restrict__ Epart,
                float* __restrict__ Zpart) {
    __shared__ __align__(16) unsigned short Pl[64 * PS];
    __shared__ __align__(16) unsigned short Vl[64 * PS];
    __shared__ float Ml[64];
    __shared__ float Zp[256];

    const int tid = threadIdx.x;
    const int h = blockIdx.x, b = blockIdx.y, s = blockIdx.z;
    const int wv = tid >> 5, ln = tid & 31, lh = ln & 15, lq = ln >> 4;
    const int dcol = tid & 63;           // this thread's d (== v) column

    // reduce the 512 per-block col-max partials of this batch (spread 4-way)
    {
        const int qq = tid >> 6;         // 0..3
        float m = -3.4e38f;
        const size_t base = ((size_t)b * 512 + qq * 128) * 512 + h * D_ + dcol;
        for (int j = 0; j < 128; ++j)
            m = fmaxf(m, Mpart[base + (size_t)j * 512]);
        Zp[tid] = m;
    }
    __syncthreads();
    if (tid < 64)
        Ml[tid] = fmaxf(fmaxf(Zp[tid], Zp[tid + 64]),
                        fmaxf(Zp[tid + 128], Zp[tid + 192]));
    __syncthreads();

    const float mreg = Ml[dcol];
    float zacc = 0.f;

    const v8f zero = {0.f, 0.f, 0.f, 0.f, 0.f, 0.f, 0.f, 0.f};
    v8f acc[2] = {zero, zero};
    const int d0 = (wv & 3) * 16;
    const int v0 = (wv >> 2) * 32;

    for (int cn = 0; cn < 16; ++cn) {
        __syncthreads();
        const size_t gbase =
            ((size_t)b * N_ + (size_t)s * 2048 + (size_t)cn * 128) * C_ + h * D_ + dcol;
#pragma unroll 4
        for (int j = 0; j < 32; ++j) {
            const int i = (tid >> 6) + j * 4;
            const float kv = kh[gbase + (size_t)i * C_];
            const float e = __expf(kv - mreg);
            zacc += e;
            Pl[dcol * PS + i] = f32_to_bf16(e);
            Vl[dcol * PS + i] = vh[gbase + (size_t)i * C_];
        }
        __syncthreads();
#pragma unroll
        for (int ks = 0; ks < 4; ++ks) {
            const int nk = ks * 32;
            Frag16 a;
            {
                const unsigned short* p = &Pl[(d0 + lh) * PS + nk + lq * 16];
                a.q[0] = *(const v4u*)(p);
                a.q[1] = *(const v4u*)(p + 8);
            }
#pragma unroll
            for (int t = 0; t < 2; ++t) {
                Frag16 bfr;
                const unsigned short* p =
                    &Vl[(v0 + t * 16 + lh) * PS + nk + lq * 16];
                bfr.q[0] = *(const v4u*)(p);
                bfr.q[1] = *(const v4u*)(p + 8);
                acc[t] = wmma_bf16(a, bfr, acc[t]);
            }
        }
    }

    // partial Z
    Zp[tid] = zacc;
    __syncthreads();
    const size_t part = (size_t)(b * H_ + h) * NS + s;
    if (tid < 64)
        Zpart[part * D_ + tid] = Zp[tid] + Zp[tid + 64] + Zp[tid + 128] + Zp[tid + 192];

    // partial E (f32, deterministic per-split buffers)
    float* ep = Epart + part * (D_ * D_);
#pragma unroll
    for (int t = 0; t < 2; ++t) {
#pragma unroll
        for (int r = 0; r < 8; ++r) {
            const int dd = d0 + r + lq * 8;
            const int vv = v0 + t * 16 + lh;
            ep[dd * D_ + vv] = acc[t][r];
        }
    }
}

// ---------------------------------------------------------------------------
// Kernel 2b: sum the NS partials, normalize by Z, store transposed bf16 ktvT.
// ---------------------------------------------------------------------------
__global__ __launch_bounds__(256) void
ktv_finalize_kernel(const float* __restrict__ Epart,
                    const float* __restrict__ Zpart,
                    unsigned short* __restrict__ ktvT) {
    __shared__ float Zl[64];
    const int bh = blockIdx.x;
    const int tid = threadIdx.x;
    if (tid < 64) {
        float z = 0.f;
        for (int s = 0; s < NS; ++s)
            z += Zpart[((size_t)bh * NS + s) * D_ + tid];
        Zl[tid] = z;
    }
    __syncthreads();
    for (int e = tid; e < D_ * D_; e += 256) {
        const int d = e >> 6, v = e & 63;
        float sum = 0.f;
        for (int s = 0; s < NS; ++s)
            sum += Epart[((size_t)bh * NS + s) * (D_ * D_) + e];
        ktvT[(size_t)bh * (D_ * D_) + v * D_ + d] = f32_to_bf16(sum / Zl[d]);
    }
}

// ---------------------------------------------------------------------------
// Kernel 3: fused factor_att + output projection.
// Block = 32 rows. Phase 1: wave wv computes head h=wv of the fa tile
// (32x64, K=64 vs ktvT) -> bf16 into LDS (stride 520, 16B aligned).
// Phase 2: out[32,512] = fa_tile @ Wp.T + bias, A-frags straight from LDS.
// Eliminates the 128 MB fa round-trip entirely. grid = (M/32).
// ---------------------------------------------------------------------------
constexpr int FS = 520;                  // fa-tile LDS row stride (elements)

__global__ __launch_bounds__(256) void
factor_out_kernel(const unsigned short* __restrict__ qh,
                  const unsigned short* __restrict__ ktvT,
                  const unsigned short* __restrict__ Wbf,
                  const float* __restrict__ bias,
                  float* __restrict__ out) {
    __shared__ __align__(16) unsigned short Fl[32 * FS];

    const int tid = threadIdx.x;
    const int wv = tid >> 5, ln = tid & 31, lh = ln & 15, lq = ln >> 4;
    const int mblk = blockIdx.x * 32;
    const int b = mblk >> 14;                   // row-block's batch (N_=16384)

    const v8f zero = {0.f, 0.f, 0.f, 0.f, 0.f, 0.f, 0.f, 0.f};

    // ---- phase 1: this wave computes head h = wv of the fa tile ----
    {
        const int h = wv;
        const unsigned short* kt = ktvT + ((size_t)(b * H_ + h)) * D_ * D_;
        v8f facc[2][4] = {{zero, zero, zero, zero}, {zero, zero, zero, zero}};
#pragma unroll
        for (int ks = 0; ks < D_; ks += 32) {
            Frag16 a[2];
#pragma unroll
            for (int mtt = 0; mtt < 2; ++mtt) {
                const unsigned short* ar =
                    qh + (size_t)(mblk + mtt * 16 + lh) * C_ + h * D_ + ks + lq * 16;
                a[mtt].q[0] = *(const v4u*)(ar);
                a[mtt].q[1] = *(const v4u*)(ar + 8);
            }
#pragma unroll
            for (int t = 0; t < 4; ++t) {
                Frag16 bfr;
                const unsigned short* p = kt + (t * 16 + lh) * D_ + ks + lq * 16;
                bfr.q[0] = *(const v4u*)(p);
                bfr.q[1] = *(const v4u*)(p + 8);
#pragma unroll
                for (int mtt = 0; mtt < 2; ++mtt)
                    facc[mtt][t] = wmma_bf16(a[mtt], bfr, facc[mtt][t]);
            }
        }
        // fa tile -> LDS (bf16)
#pragma unroll
        for (int mtt = 0; mtt < 2; ++mtt)
#pragma unroll
            for (int t = 0; t < 4; ++t) {
                const int col = h * D_ + t * 16 + lh;
#pragma unroll
                for (int r = 0; r < 8; ++r) {
                    const int row = mtt * 16 + r + lq * 8;
                    Fl[row * FS + col] = f32_to_bf16(facc[mtt][t][r]);
                }
            }
    }
    __syncthreads();

    // ---- phase 2: out = fa @ Wp.T + bias ----
    const int mt = wv & 1, nt = wv >> 1;
    const int m0 = mblk + mt * 16;
    const int n0 = nt * 128;
    v8f acc[8] = {zero, zero, zero, zero, zero, zero, zero, zero};

    for (int ks = 0; ks < 16; ++ks) {
        const int k0 = ks * 32;
        Frag16 a;
        {
            const unsigned short* ap = &Fl[(mt * 16 + lh) * FS + k0 + lq * 16];
            a.q[0] = *(const v4u*)(ap);
            a.q[1] = *(const v4u*)(ap + 8);
        }
#pragma unroll
        for (int t = 0; t < 8; ++t) {
            Frag16 bfr;
            const unsigned short* br =
                Wbf + (size_t)(n0 + t * 16 + lh) * C_ + k0 + lq * 16;
            bfr.q[0] = *(const v4u*)(br);
            bfr.q[1] = *(const v4u*)(br + 8);
            acc[t] = wmma_bf16(a, bfr, acc[t]);
        }
    }
#pragma unroll
    for (int t = 0; t < 8; ++t) {
        const int col = n0 + t * 16 + lh;
        const float bv = bias[col];
#pragma unroll
        for (int r = 0; r < 8; ++r) {
            const size_t row = (size_t)(m0 + r + lq * 8);
            out[row * C_ + col] = acc[t][r] + bv;
        }
    }
}

// ---------------------------------------------------------------------------
// Launch
// ---------------------------------------------------------------------------
extern "C" void kernel_launch(void* const* d_in, const int* in_sizes, int n_in,
                              void* d_out, int out_size, void* d_ws, size_t ws_size,
                              hipStream_t stream) {
    const float* q  = (const float*)d_in[0];
    const float* k  = (const float*)d_in[1];
    const float* v  = (const float*)d_in[2];
    const float* Wq = (const float*)d_in[3];
    const float* Wk = (const float*)d_in[4];
    const float* Wv = (const float*)d_in[5];
    const float* Wp = (const float*)d_in[6];
    const float* bp = (const float*)d_in[7];
    float* out = (float*)d_out;
    char* ws = (char*)d_ws;

    unsigned short* Wqb  = (unsigned short*)(ws + OFF_WQ);
    unsigned short* Wkb  = (unsigned short*)(ws + OFF_WK);
    unsigned short* Wvb  = (unsigned short*)(ws + OFF_WV);
    unsigned short* Wpb  = (unsigned short*)(ws + OFF_WP);
    unsigned short* qh   = (unsigned short*)(ws + OFF_QH);
    unsigned short* vhp  = (unsigned short*)(ws + OFF_VH);
    float*          khp  = (float*)         (ws + OFF_KH);
    float*          Mp   = (float*)         (ws + OFF_MP);
    float*          Ep   = (float*)         (ws + OFF_EP);
    float*          Zp   = (float*)         (ws + OFF_ZP);
    unsigned short* ktvT = (unsigned short*)(ws + OFF_KTVT);

    const int wn = C_ * C_;                       // 262144 per weight
    cvt_bf16_kernel<<<wn / 256, 256, 0, stream>>>(Wq, Wqb, wn);
    cvt_bf16_kernel<<<wn / 256, 256, 0, stream>>>(Wk, Wkb, wn);
    cvt_bf16_kernel<<<wn / 256, 256, 0, stream>>>(Wv, Wvb, wn);
    cvt_bf16_kernel<<<wn / 256, 256, 0, stream>>>(Wp, Wpb, wn);

    dim3 gemm_grid(NBLK);
    gemm_xwt_kernel<true,  false><<<gemm_grid, 256, 0, stream>>>(q, Wqb, qh,  nullptr);
    gemm_xwt_kernel<false, true ><<<gemm_grid, 256, 0, stream>>>(k, Wkb, khp, Mp);
    gemm_xwt_kernel<true,  false><<<gemm_grid, 256, 0, stream>>>(v, Wvb, vhp, nullptr);

    ktv_part_kernel<<<dim3(H_, B_, NS), 256, 0, stream>>>(khp, vhp, Mp, Ep, Zp);
    ktv_finalize_kernel<<<B_ * H_, 256, 0, stream>>>(Ep, Zp, ktvT);

    factor_out_kernel<<<gemm_grid, 256, 0, stream>>>(qh, ktvT, Wpb, bp, out);
}
